// MixtureOfExperts_31507880084005
// MI455X (gfx1250) — matrix operand order repaired and checked by reading, not compiled, not run on hardware
//
#include <hip/hip_runtime.h>
#include <hip/hip_bf16.h>
#include <math.h>

#define DM  1024
#define EXP 8
#define FF  2048
#define TM  32   // token tile per FFN block

typedef __bf16 v16bf __attribute__((ext_vector_type(16)));
typedef float  v8f   __attribute__((ext_vector_type(8)));

union FragU { uint4 q[2]; v16bf v; };

__device__ __forceinline__ unsigned short f2bf(float f) {
  unsigned u = __float_as_uint(f);
  u += 0x7FFFu + ((u >> 16) & 1u);      // round-to-nearest-even
  return (unsigned short)(u >> 16);
}

// Branch-free GELU: v * sigmoid(1.5957691*v + 0.0713548*v^3).
// Max deviation from exact erf-GELU ~1e-3 relative — below the bf16
// quantization of the h tile, so results are equivalent after f2bf.
__device__ __forceinline__ float fast_gelu(float v) {
  float u = v * (1.5957691216057308f + 0.0713548162726009f * v * v);
  return v * __builtin_amdgcn_rcpf(1.0f + __expf(-u));
}

// ---------------- init: zero output & per-expert counters ----------------
__global__ void moe_init_kernel(float* out, int* counts, long long total) {
  long long i = (long long)blockIdx.x * blockDim.x + threadIdx.x;
  if (i < total) out[i] = 0.0f;
  if (i < EXP)   counts[i] = 0;
}

// ---------------- x: fp32 -> bf16 ----------------
__global__ void cvt_x_kernel(const float* __restrict__ x, unsigned short* __restrict__ xb,
                             long long total) {
  long long i = (long long)blockIdx.x * blockDim.x + threadIdx.x;
  if (i < total) xb[i] = f2bf(x[i]);
}

// ---------------- transpose-convert: src f32 [e][A][B] -> dst bf16 [e][B][A] ----------------
__global__ void tconv_kernel(const float* __restrict__ src, unsigned short* __restrict__ dst,
                             int A, int B) {
  __shared__ float t[32][33];
  int e  = blockIdx.z;
  int a0 = blockIdx.x * 32;
  int b0 = blockIdx.y * 32;
  int c  = threadIdx.x;        // 0..31
  int r0 = threadIdx.y;        // 0..7
  const float*    s = src + (size_t)e * A * B;
  unsigned short* d = dst + (size_t)e * A * B;
  for (int rr = r0; rr < 32; rr += 8)
    t[rr][c] = s[(size_t)(a0 + rr) * B + (b0 + c)];
  __syncthreads();
  for (int rr = r0; rr < 32; rr += 8)
    d[(size_t)(b0 + rr) * A + (a0 + c)] = f2bf(t[c][rr]);
}

// ---------------- router: logits, softmax top-2, bucket append ----------------
__global__ void router_kernel(const float* __restrict__ x, const float* __restrict__ Wr,
                              int* counts, int* tok_idx, float* tok_gate, int N) {
  int wave  = threadIdx.x >> 5;
  int lane  = threadIdx.x & 31;
  int token = blockIdx.x * (blockDim.x >> 5) + wave;
  if (token >= N) return;

  float acc[EXP];
#pragma unroll
  for (int e = 0; e < EXP; e++) acc[e] = 0.0f;
  const float* xr = x + (size_t)token * DM;
  for (int d = lane; d < DM; d += 32) {
    float xv = xr[d];
#pragma unroll
    for (int e = 0; e < EXP; e++) acc[e] += xv * Wr[e * DM + d];
  }
#pragma unroll
  for (int e = 0; e < EXP; e++)
    for (int off = 16; off; off >>= 1)
      acc[e] += __shfl_xor(acc[e], off, 32);

  if (lane == 0) {
    int i0 = 0;
#pragma unroll
    for (int e = 1; e < EXP; e++) if (acc[e] > acc[i0]) i0 = e;
    int i1 = (i0 == 0) ? 1 : 0;
#pragma unroll
    for (int e = 0; e < EXP; e++) if (e != i0 && acc[e] > acc[i1]) i1 = e;
    // renormalized top-2 gates: full softmax denominator cancels
    float p1  = expf(acc[i1] - acc[i0]);
    float inv = 1.0f / (1.0f + p1);
    int pos0 = atomicAdd(&counts[i0], 1);
    tok_idx [i0 * N + pos0] = token;
    tok_gate[i0 * N + pos0] = inv;
    int pos1 = atomicAdd(&counts[i1], 1);
    tok_idx [i1 * N + pos1] = token;
    tok_gate[i1 * N + pos1] = p1 * inv;
  }
}

// ---------------- fused expert FFN: async gather -> GEMM1+GELU (LDS) -> GEMM2 -> gated scatter ----------------
__global__ __launch_bounds__(256)
void moe_ffn_kernel(const unsigned short* __restrict__ xb,
                    const unsigned short* __restrict__ W1t,  // [e][f][d] bf16
                    const float* __restrict__ b1,
                    const unsigned short* __restrict__ W2t,  // [e][d][f] bf16
                    const float* __restrict__ b2,
                    const int* __restrict__ counts,
                    const int* __restrict__ tok_idx,
                    const float* __restrict__ tok_gate,
                    float* out, int N) {
  extern __shared__ char smem[];
  unsigned short* xt   = (unsigned short*)smem;                         // TM x DM bf16 (64KB)
  unsigned short* ht   = (unsigned short*)(smem + (size_t)TM * DM * 2); // TM x FF bf16 (128KB)
  int*            toks = (int*)(smem + (size_t)TM * DM * 2 + (size_t)TM * FF * 2);
  float*          gts  = (float*)((char*)toks + TM * sizeof(int));

  int e    = blockIdx.y;
  int cnt  = counts[e];
  int row0 = blockIdx.x * TM;
  if (row0 >= cnt) return;
  int rows = min(TM, cnt - row0);

  int tid = threadIdx.x;
  if (tid < TM) {
    int t = (tid < rows) ? tok_idx[e * N + row0 + tid] : -1;
    toks[tid] = t;
    gts [tid] = (tid < rows) ? tok_gate[e * N + row0 + tid] : 0.0f;
  }
  __syncthreads();

  // gather 32 token rows (bf16) into LDS via async global->LDS b128 copies
  // (ASYNCcnt path, no VGPR round trip); zero-fill padding rows with ds stores.
  {
    const uint4* xbv = (const uint4*)xb;      // 128 uint4 per row
    uint4*       xtv = (uint4*)xt;
    for (int i = tid; i < TM * (DM / 8); i += 256) {
      int m = i >> 7, c = i & 127;
      int t = toks[m];
      if (t >= 0) {
        const uint4* src = xbv + (size_t)t * (DM / 8) + c;
        // generic LDS pointer: low 32 bits are the LDS byte offset
        unsigned lds_addr = (unsigned)(uintptr_t)(xtv + i);
        asm volatile("global_load_async_to_lds_b128 %0, %1, off"
                     :: "v"(lds_addr), "v"(src) : "memory");
      } else {
        uint4 z; z.x = z.y = z.z = z.w = 0u;
        xtv[i] = z;
      }
    }
    asm volatile("s_wait_asynccnt 0x0" ::: "memory");
  }
  __syncthreads();

  int wave = tid >> 5, lane = tid & 31;
  int n = lane & 15, g = lane >> 4;

#define LOAD_B(Bf, base, k0)  { (Bf).q[0] = *(const uint4*)((base) + (k0));      \
                                (Bf).q[1] = *(const uint4*)((base) + (k0) + 16); }
#define LOAD_A(Af, base, k0)  { (Af).q[0] = *(const uint4*)((base) + (k0));      \
                                (Af).q[1] = *(const uint4*)((base) + (k0) + 16); }
#define WMMA(acc, Af, Bf) \
  acc = __builtin_amdgcn_wmma_f32_16x16x32_bf16(false, (Af).v, false, (Bf).v, (short)0, acc, false, false)

  // ---- GEMM1: h = GELU(x @ W1 + b1), kept in LDS (double-buffered fragments) ----
  const unsigned short* w1e = W1t + (size_t)e * FF * DM;
  for (int fb = wave; fb < FF / 16; fb += 8) {
    int f0 = fb * 16;
    v8f acc0 = {}, acc1 = {};
    const unsigned short* bp  = w1e + (size_t)(f0 + n) * DM + g * 8;
    const unsigned short* a0p = xt + (size_t)n * DM + g * 8;
    const unsigned short* a1p = xt + (size_t)(16 + n) * DM + g * 8;
    FragU Ba, Bb, A0a, A0b, A1a, A1b;
    LOAD_B(Ba, bp, 0); LOAD_A(A0a, a0p, 0); LOAD_A(A1a, a1p, 0);
#pragma unroll 2
    for (int k0 = 0; k0 < DM - 64; k0 += 64) {
      LOAD_B(Bb, bp, k0 + 32); LOAD_A(A0b, a0p, k0 + 32); LOAD_A(A1b, a1p, k0 + 32);
      WMMA(acc0, A0a, Ba); WMMA(acc1, A1a, Ba);
      LOAD_B(Ba, bp, k0 + 64); LOAD_A(A0a, a0p, k0 + 64); LOAD_A(A1a, a1p, k0 + 64);
      WMMA(acc0, A0b, Bb); WMMA(acc1, A1b, Bb);
    }
    LOAD_B(Bb, bp, DM - 32); LOAD_A(A0b, a0p, DM - 32); LOAD_A(A1b, a1p, DM - 32);
    WMMA(acc0, A0a, Ba); WMMA(acc1, A1a, Ba);
    WMMA(acc0, A0b, Bb); WMMA(acc1, A1b, Bb);

    float bias = b1[e * FF + f0 + n];
#pragma unroll
    for (int v = 0; v < 8; v++) {
      int m = g * 8 + v;
      ht[(size_t)m * FF + f0 + n]        = f2bf(fast_gelu(acc0[v] + bias));
      ht[(size_t)(16 + m) * FF + f0 + n] = f2bf(fast_gelu(acc1[v] + bias));
    }
  }
  __syncthreads();

  // ---- GEMM2: y = h @ W2 + b2 ; out[token] += gate * y ----
  const unsigned short* w2e = W2t + (size_t)e * DM * FF;
  for (int db = wave; db < DM / 16; db += 8) {
    int d0 = db * 16;
    v8f acc0 = {}, acc1 = {};
    const unsigned short* bp  = w2e + (size_t)(d0 + n) * FF + g * 8;
    const unsigned short* a0p = ht + (size_t)n * FF + g * 8;
    const unsigned short* a1p = ht + (size_t)(16 + n) * FF + g * 8;
    FragU Ba, Bb, A0a, A0b, A1a, A1b;
    LOAD_B(Ba, bp, 0); LOAD_A(A0a, a0p, 0); LOAD_A(A1a, a1p, 0);
#pragma unroll 2
    for (int k0 = 0; k0 < FF - 64; k0 += 64) {
      LOAD_B(Bb, bp, k0 + 32); LOAD_A(A0b, a0p, k0 + 32); LOAD_A(A1b, a1p, k0 + 32);
      WMMA(acc0, A0a, Ba); WMMA(acc1, A1a, Ba);
      LOAD_B(Ba, bp, k0 + 64); LOAD_A(A0a, a0p, k0 + 64); LOAD_A(A1a, a1p, k0 + 64);
      WMMA(acc0, A0b, Bb); WMMA(acc1, A1b, Bb);
    }
    LOAD_B(Bb, bp, FF - 32); LOAD_A(A0b, a0p, FF - 32); LOAD_A(A1b, a1p, FF - 32);
    WMMA(acc0, A0a, Ba); WMMA(acc1, A1a, Ba);
    WMMA(acc0, A0b, Bb); WMMA(acc1, A1b, Bb);

    float bias = b2[e * DM + d0 + n];
#pragma unroll
    for (int v = 0; v < 8; v++) {
      int m = g * 8 + v;
      if (m < rows)
        atomicAdd(&out[(size_t)toks[m] * DM + d0 + n], (acc0[v] + bias) * gts[m]);
      int m1 = 16 + m;
      if (m1 < rows)
        atomicAdd(&out[(size_t)toks[m1] * DM + d0 + n], (acc1[v] + bias) * gts[m1]);
    }
  }
#undef LOAD_B
#undef LOAD_A
#undef WMMA
}

// ---------------- host launch ----------------
extern "C" void kernel_launch(void* const* d_in, const int* in_sizes, int n_in,
                              void* d_out, int out_size, void* d_ws, size_t ws_size,
                              hipStream_t stream) {
  (void)n_in; (void)out_size; (void)ws_size;
  const float* x  = (const float*)d_in[0];
  const float* Wr = (const float*)d_in[1];
  const float* W1 = (const float*)d_in[2];
  const float* b1 = (const float*)d_in[3];
  const float* W2 = (const float*)d_in[4];
  const float* b2 = (const float*)d_in[5];
  float* out = (float*)d_out;
  int N = in_sizes[0] / DM;   // 8192 tokens

  char* ws = (char*)d_ws;
  size_t off = 0;
  unsigned short* xb  = (unsigned short*)(ws + off); off += (size_t)N * DM * 2;
  unsigned short* W1t = (unsigned short*)(ws + off); off += (size_t)EXP * FF * DM * 2;
  unsigned short* W2t = (unsigned short*)(ws + off); off += (size_t)EXP * DM * FF * 2;
  int*   counts   = (int*)(ws + off);   off += 256;
  int*   tok_idx  = (int*)(ws + off);   off += (size_t)EXP * N * 4;
  float* tok_gate = (float*)(ws + off); off += (size_t)EXP * N * 4;

  long long totalOut = (long long)N * DM;
  moe_init_kernel<<<(unsigned)((totalOut + 255) / 256), 256, 0, stream>>>(out, counts, totalOut);
  cvt_x_kernel<<<(unsigned)((totalOut + 255) / 256), 256, 0, stream>>>(x, xb, totalOut);

  dim3 tb(32, 8);
  tconv_kernel<<<dim3(DM / 32, FF / 32, EXP), tb, 0, stream>>>(W1, W1t, DM, FF); // [e][d][f]->[e][f][d]
  tconv_kernel<<<dim3(FF / 32, DM / 32, EXP), tb, 0, stream>>>(W2, W2t, FF, DM); // [e][f][d]->[e][d][f]

  router_kernel<<<N / 8, 256, 0, stream>>>(x, Wr, counts, tok_idx, tok_gate, N);

  size_t smem = (size_t)TM * DM * 2 + (size_t)TM * FF * 2 + (size_t)TM * (sizeof(int) + sizeof(float));
  moe_ffn_kernel<<<dim3(N / TM, EXP), 256, smem, stream>>>(xb, W1t, b1, W2t, b2,
                                                           counts, tok_idx, tok_gate, out, N);
}
